// MyCLRHead_27728308863120
// MI455X (gfx1250) — matrix operand order repaired and checked by reading, not compiled, not run on hardware
//
#include <hip/hip_runtime.h>
#include <math.h>

typedef __attribute__((ext_vector_type(16))) _Float16 v16h;
typedef __attribute__((ext_vector_type(8)))  _Float16 v8h;
typedef __attribute__((ext_vector_type(8)))  float    v8f;

#define DEV __device__ __forceinline__

#define IMG_W_ 800
#define IMG_H_ 320
#define NPT 72
#define S_ 36
#define P_ 192
#define C64 64
#define B_ 32
#define BP 6144   // B_*P_
#define PI_F 3.14159265358979323846f

DEV int sx_of(int s) { return (s * 71) / 35; }                 // SAMPLE_X_IDX
DEV float feat_y(int s) { return 1.0f - (float)sx_of(35 - s) / 71.0f; }  // PRIOR_FEAT_YS

DEV v8f vzero8() { v8f v = {0.f,0.f,0.f,0.f,0.f,0.f,0.f,0.f}; return v; }
DEV v8h vzero8h() { v8h v;
#pragma unroll
  for (int i = 0; i < 8; ++i) v[i] = (_Float16)0.0f;
  return v; }

DEV v8f wmma16(v16h a, v16h b, v8f c) {
  return __builtin_amdgcn_wmma_f32_16x16x32_f16(false, a, false, b, (short)0, c, false, false);
}

// ISA 16-bit A-matrix 16x32 layout: lane<16 holds K {0..7,16..23}, lane>=16 holds K {8..15,24..31}
DEV int kA(int lane, int e) { return ((lane >> 4) << 3) + e + ((e >= 8) ? 8 : 0); }

DEV v16h mk16(v8h lo, v8h hi) {
  v16h r;
#pragma unroll
  for (int i = 0; i < 8; ++i) { r[i] = lo[i]; r[i + 8] = hi[i]; }
  return r;
}
DEV v8h  ld8(const _Float16* p)  { return *(const v8h*)p; }
DEV v16h ld16(const _Float16* p) { return *(const v16h*)p; }

// ---------------- grid_sample pooling:  pooled[(n*36+s)*64+c] (channels-last f16) -------------
__global__ void k_pool(const float* __restrict__ fmap, int h, int w,
                       const float* __restrict__ pfeat, _Float16* __restrict__ out) {
  int blk = blockIdx.x;            // n*36 + s
  int n = blk / S_;
  int s = blk - n * S_;
  int b = n / P_;
  int c = threadIdx.x;             // 0..63
  float pf = pfeat[(size_t)n * S_ + (S_ - 1 - s)];   // reversed prior_xs
  float x = pf * (float)(w - 1);
  float y = feat_y(s) * (float)(h - 1);
  float x0f = floorf(x), y0f = floorf(y);
  int x0 = (int)x0f, y0 = (int)y0f;
  float wx = x - x0f, wy = y - y0f;
  const float* img = fmap + (size_t)(b * C64 + c) * h * w;
  float acc = 0.f;
#pragma unroll
  for (int dy = 0; dy < 2; ++dy)
#pragma unroll
    for (int dx = 0; dx < 2; ++dx) {
      int xi = x0 + dx, yi = y0 + dy;
      bool valid = (xi >= 0) && (xi <= w - 1) && (yi >= 0) && (yi <= h - 1);
      int xc = xi < 0 ? 0 : (xi > w - 1 ? w - 1 : xi);
      int yc = yi < 0 ? 0 : (yi > h - 1 ? h - 1 : yi);
      float v = valid ? img[yc * w + xc] : 0.f;
      float wgt = (dx ? wx : 1.f - wx) * (dy ? wy : 1.f - wy);
      acc += v * wgt;
    }
  out[(size_t)blk * C64 + c] = (_Float16)acc;
}

DEV void conv_store(const v8f* acc, int n, int s0, int m, int hi, const float* bn,
                    _Float16* out, int out_mode, int out_choff, int out_stride) {
#pragma unroll
  for (int nt = 0; nt < 4; ++nt) {
    int co = nt * 16 + m;
    float g = bn[co], bb = bn[64 + co], mn = bn[128 + co], vr = bn[192 + co];
    float sc = g * rsqrtf(vr + 1e-5f);
#pragma unroll
    for (int r = 0; r < 8; ++r) {
      int row = s0 + r + 8 * hi;
      if (row < S_) {
        float y = fmaxf((acc[nt][r] - mn) * sc + bb, 0.f);
        if (out_mode == 0)
          out[((size_t)n * S_ + row) * out_stride + out_choff + co] = (_Float16)y;
        else
          out[((size_t)n * 64 + co) * S_ + row] = (_Float16)y;
      }
    }
  }
}

// ---------------- conv1d(k=9,pad=4) as implicit-GEMM WMMA, fused BN+ReLU ----------------------
// Each wave: 2 ROIs x 16 rows x 64 cols; B fragments shared -> 12 b128 loads per 8 WMMAs.
__global__ void __launch_bounds__(32) k_conv(
    const _Float16* __restrict__ in, int CinDiv32, int in_stride,
    const _Float16* __restrict__ wfrag, const float* __restrict__ bn,
    _Float16* __restrict__ out, int out_mode, int out_choff, int out_stride) {
  int n0 = blockIdx.x * 2;
  int tile = blockIdx.y;           // 0..2  (rows s0..s0+15)
  int lane = threadIdx.x & 31;
  int s0 = tile * 16;
  int m = lane & 15;
  int hi = lane >> 4;
  int kb = hi * 8;
  v8f acc0[4], acc1[4];
#pragma unroll
  for (int t = 0; t < 4; ++t) { acc0[t] = vzero8(); acc1[t] = vzero8(); }
  const _Float16* inr0 = in + (size_t)n0 * S_ * in_stride;
  const _Float16* inr1 = inr0 + (size_t)S_ * in_stride;
  const v8h z8 = vzero8h();
  int kt = 0;
  for (int kk = 0; kk < 9; ++kk) {
    int sp = s0 + m + kk - 4;
    bool valid = (sp >= 0) && (sp < S_);
    size_t ro = (size_t)(valid ? sp : 0) * in_stride;
    for (int cb = 0; cb < CinDiv32; ++cb, ++kt) {
      int ci = cb * 32 + kb;
      v8h a0l = ld8(inr0 + ro + ci);
      v8h a0h = ld8(inr0 + ro + ci + 16);
      v8h a1l = ld8(inr1 + ro + ci);
      v8h a1h = ld8(inr1 + ro + ci + 16);
      const _Float16* wp = wfrag + (size_t)kt * 2048 + (size_t)lane * 16;
      v16h b0 = ld16(wp);
      v16h b1 = ld16(wp + 512);
      v16h b2 = ld16(wp + 1024);
      v16h b3 = ld16(wp + 1536);
      __builtin_prefetch((const void*)(wp + 2048), 0, 3);
      a0l = valid ? a0l : z8;  a0h = valid ? a0h : z8;
      a1l = valid ? a1l : z8;  a1h = valid ? a1h : z8;
      v16h a0 = mk16(a0l, a0h);
      v16h a1 = mk16(a1l, a1h);
      acc0[0] = wmma16(a0, b0, acc0[0]);
      acc1[0] = wmma16(a1, b0, acc1[0]);
      acc0[1] = wmma16(a0, b1, acc0[1]);
      acc1[1] = wmma16(a1, b1, acc1[1]);
      acc0[2] = wmma16(a0, b2, acc0[2]);
      acc1[2] = wmma16(a1, b2, acc1[2]);
      acc0[3] = wmma16(a0, b3, acc0[3]);
      acc1[3] = wmma16(a1, b3, acc1[3]);
    }
  }
  conv_store(acc0, n0,     s0, m, hi, bn, out, out_mode, out_choff, out_stride);
  conv_store(acc1, n0 + 1, s0, m, hi, bn, out, out_mode, out_choff, out_stride);
}

// ---------------- FC (2304->64) + bias + LayerNorm + ReLU;  also emit scaled query ------------
// Each wave: 32 rows x 64 cols, shared B fragments; all 32 lanes active in LN epilogue.
__global__ void __launch_bounds__(32) k_fc(
    const _Float16* __restrict__ catA, const _Float16* __restrict__ wfrag,
    const float* __restrict__ fc_b, const float* __restrict__ ln_g,
    const float* __restrict__ ln_b, const float* __restrict__ fq_w,
    const float* __restrict__ fq_b,
    float* __restrict__ roi_f, _Float16* __restrict__ query_h) {
  __shared__ float ts[32 * 64];
  int n0 = blockIdx.x * 32;
  int lane = threadIdx.x & 31;
  int m = lane & 15, hi = lane >> 4, kb = hi * 8;
  v8f acc0[4], acc1[4];
#pragma unroll
  for (int t = 0; t < 4; ++t) { acc0[t] = vzero8(); acc1[t] = vzero8(); }
  const _Float16* ar0 = catA + (size_t)(n0 + m) * 2304;
  const _Float16* ar1 = ar0 + (size_t)16 * 2304;
  for (int kt = 0; kt < 72; ++kt) {
    int q0 = kt * 32 + kb;
    v16h a0 = mk16(ld8(ar0 + q0), ld8(ar0 + q0 + 16));
    v16h a1 = mk16(ld8(ar1 + q0), ld8(ar1 + q0 + 16));
    const _Float16* wp = wfrag + (size_t)kt * 2048 + (size_t)lane * 16;
    v16h b0 = ld16(wp);
    v16h b1 = ld16(wp + 512);
    v16h b2 = ld16(wp + 1024);
    v16h b3 = ld16(wp + 1536);
    __builtin_prefetch((const void*)(wp + 2048), 0, 3);
    acc0[0] = wmma16(a0, b0, acc0[0]);
    acc1[0] = wmma16(a1, b0, acc1[0]);
    acc0[1] = wmma16(a0, b1, acc0[1]);
    acc1[1] = wmma16(a1, b1, acc1[1]);
    acc0[2] = wmma16(a0, b2, acc0[2]);
    acc1[2] = wmma16(a1, b2, acc1[2]);
    acc0[3] = wmma16(a0, b3, acc0[3]);
    acc1[3] = wmma16(a1, b3, acc1[3]);
  }
#pragma unroll
  for (int nt = 0; nt < 4; ++nt) {
    int col = nt * 16 + m;
    float bias = fc_b[col];
#pragma unroll
    for (int r = 0; r < 8; ++r) {
      int row = r + 8 * hi;
      ts[row * 64 + col] = acc0[nt][r] + bias;
      ts[(row + 16) * 64 + col] = acc1[nt][r] + bias;
    }
  }
  __syncthreads();
  {
    int row = lane, n = n0 + row, p = n % P_;
    float mu = 0.f;
    for (int c = 0; c < 64; ++c) mu += ts[row * 64 + c];
    mu *= (1.f / 64.f);
    float var = 0.f;
    for (int c = 0; c < 64; ++c) { float d = ts[row * 64 + c] - mu; var += d * d; }
    var *= (1.f / 64.f);
    float inv = rsqrtf(var + 1e-5f);
    float fw = fq_w[p], fb = fq_b[p];
    for (int c = 0; c < 64; ++c) {
      float v = fmaxf((ts[row * 64 + c] - mu) * inv * ln_g[c] + ln_b[c], 0.f);
      roi_f[(size_t)n * 64 + c] = v;
      float qv = fmaxf(v * fw + fb, 0.f) * 0.125f;   // fold C^-0.5 into query
      query_h[(size_t)n * 64 + c] = (_Float16)qv;
    }
  }
}

// ---------------- key/value 1x1 conv at the 250 resized pixels, fused BN/ReLU & bias ----------
// key stored transposed (b,256,64) [pix][c]; value stored (b,64,256) [c][pix]; pads zeroed.
__global__ void __launch_bounds__(32) k_kv(
    const float* __restrict__ fmap, int h, int w,
    const _Float16* __restrict__ wfrag, const float* __restrict__ bn_key,
    const float* __restrict__ fval_b,
    _Float16* __restrict__ key_t, _Float16* __restrict__ value_t) {
  int tile = blockIdx.x;           // 0..15 -> pixels tile*16..+15 (pad to 256)
  int b = blockIdx.y;
  int lane = threadIdx.x & 31;
  int m = lane & 15, hi = lane >> 4;
  v8f acc[8];
#pragma unroll
  for (int t = 0; t < 8; ++t) acc[t] = vzero8();
  for (int kt = 0; kt < 2; ++kt) {
    v16h a;
#pragma unroll
    for (int e = 0; e < 16; ++e) {
      int c = kt * 32 + kA(lane, e);
      int pix = tile * 16 + m;
      float v = 0.f;
      if (pix < 250) {
        int i = pix / 25, j = pix - 25 * i;
        int y = i * h / 10, x = j * w / 25;
        v = fmap[(((size_t)b * C64 + c) * h + y) * w + x];
      }
      a[e] = (_Float16)v;
    }
    const _Float16* wp = wfrag + (size_t)kt * 8 * 512 + (size_t)lane * 16;
#pragma unroll
    for (int nt = 0; nt < 8; ++nt)
      acc[nt] = wmma16(a, ld16(wp + (size_t)nt * 512), acc[nt]);
  }
#pragma unroll
  for (int nt = 0; nt < 8; ++nt) {
    int col = nt * 16 + m;
#pragma unroll
    for (int r = 0; r < 8; ++r) {
      int pix = tile * 16 + r + 8 * hi;
      if (col < 64) {
        float g = bn_key[col], bb = bn_key[64 + col], mn = bn_key[128 + col], vr = bn_key[192 + col];
        float y = (pix < 250) ? fmaxf((acc[nt][r] - mn) * g * rsqrtf(vr + 1e-5f) + bb, 0.f) : 0.f;
        key_t[((size_t)b * 256 + pix) * 64 + col] = (_Float16)y;
      } else {
        int co = col - 64;
        float y = (pix < 250) ? (acc[nt][r] + fval_b[co]) : 0.f;   // zero pad rows for K=256 GEMM
        value_t[((size_t)b * 64 + co) * 256 + pix] = (_Float16)y;
      }
    }
  }
}

// ---------------- sim = softmax(Q K): 16 rows x 256 cols per block (4 waves) ------------------
__global__ void __launch_bounds__(128) k_attn1(
    const _Float16* __restrict__ query_h, const _Float16* __restrict__ key_t,
    _Float16* __restrict__ sim_h) {
  __shared__ float ts[16 * 256];
  int mt = blockIdx.x;             // 0..11
  int b = blockIdx.y;
  int wv = threadIdx.x >> 5;
  int lane = threadIdx.x & 31;
  int m = lane & 15, hi = lane >> 4, kb = hi * 8, kb16 = hi * 16;
  int m0 = mt * 16;
  v8f acc[4];
#pragma unroll
  for (int t = 0; t < 4; ++t) acc[t] = vzero8();
  const _Float16* qrow = query_h + ((size_t)b * P_ + m0 + m) * 64;
  for (int kt = 0; kt < 2; ++kt) {
    int q0 = kt * 32;
    v16h a = ld16(qrow + q0 + kb16);
#pragma unroll
    for (int nt = 0; nt < 4; ++nt) {
      int col = wv * 64 + nt * 16 + m;
      v16h bf = ld16(key_t + ((size_t)b * 256 + col) * 64 + q0 + kb16);
      acc[nt] = wmma16(a, bf, acc[nt]);
    }
  }
#pragma unroll
  for (int nt = 0; nt < 4; ++nt) {
    int col = wv * 64 + nt * 16 + m;
#pragma unroll
    for (int r = 0; r < 8; ++r) ts[(r + 8 * hi) * 256 + col] = acc[nt][r];
  }
  __syncthreads();
  if (threadIdx.x < 16) {
    int row = threadIdx.x;
    float mx = -1e30f;
    for (int k = 0; k < 250; ++k) mx = fmaxf(mx, ts[row * 256 + k]);
    float sum = 0.f;
    for (int k = 0; k < 250; ++k) { float e = __expf(ts[row * 256 + k] - mx); ts[row * 256 + k] = e; sum += e; }
    float inv = 1.f / sum;
    _Float16* so = sim_h + ((size_t)b * P_ + m0 + row) * 256;
    for (int k = 0; k < 250; ++k) so[k] = (_Float16)(ts[row * 256 + k] * inv);
    for (int k = 250; k < 256; ++k) so[k] = (_Float16)0.f;
  }
}

// ---------------- context = sim V; roi2 = roi + attnW*context + attnB -------------------------
__global__ void __launch_bounds__(32) k_attn2(
    const _Float16* __restrict__ sim_h, const _Float16* __restrict__ value_t,
    const float* __restrict__ roi_f, const float* __restrict__ aw,
    const float* __restrict__ ab, _Float16* __restrict__ roi2_h) {
  int mt = blockIdx.x;
  int b = blockIdx.y;
  int lane = threadIdx.x & 31;
  int m = lane & 15, hi = lane >> 4, kb = hi * 8, kb16 = hi * 16;
  int m0 = mt * 16;
  v8f acc[4];
#pragma unroll
  for (int t = 0; t < 4; ++t) acc[t] = vzero8();
  const _Float16* srow = sim_h + ((size_t)b * P_ + m0 + m) * 256;
  for (int kt = 0; kt < 8; ++kt) {
    int q0 = kt * 32;
    v16h a = ld16(srow + q0 + kb16);
#pragma unroll
    for (int nt = 0; nt < 4; ++nt) {
      int col = nt * 16 + m;
      v16h bf = ld16(value_t + ((size_t)b * 64 + col) * 256 + q0 + kb16);
      acc[nt] = wmma16(a, bf, acc[nt]);
    }
  }
#pragma unroll
  for (int nt = 0; nt < 4; ++nt) {
    int col = nt * 16 + m;
#pragma unroll
    for (int r = 0; r < 8; ++r) {
      int p = m0 + r + 8 * hi;
      size_t n = (size_t)b * P_ + p;
      float v = roi_f[n * 64 + col] + acc[nt][r] * aw[p] + ab[p];
      roi2_h[n * 64 + col] = (_Float16)v;
    }
  }
}

// ---------------- fused MLP (2x 64->64 relu) + combined cls/reg head (64->80) -----------------
__global__ void __launch_bounds__(32) k_mlp(
    const _Float16* __restrict__ roi2_h, const _Float16* __restrict__ mlpf,
    const float* __restrict__ mlp_b, const _Float16* __restrict__ headf,
    const float* __restrict__ whead_b, float* __restrict__ hd) {
  __shared__ _Float16 lt[16 * 64];
  int n0 = blockIdx.x * 16;
  int lane = threadIdx.x & 31;
  int m = lane & 15, hi = lane >> 4, kb = hi * 8;
  v8f acc[4];
  // GEMM1
#pragma unroll
  for (int t = 0; t < 4; ++t) acc[t] = vzero8();
  const _Float16* arow = roi2_h + (size_t)(n0 + m) * 64;
#pragma unroll
  for (int kt = 0; kt < 2; ++kt) {
    int q0 = kt * 32;
    v16h a = mk16(ld8(arow + q0 + kb), ld8(arow + q0 + kb + 16));
    const _Float16* wp = mlpf + (size_t)kt * 4 * 512 + (size_t)lane * 16;
#pragma unroll
    for (int nt = 0; nt < 4; ++nt)
      acc[nt] = wmma16(a, ld16(wp + (size_t)nt * 512), acc[nt]);
  }
#pragma unroll
  for (int nt = 0; nt < 4; ++nt) {
    int col = nt * 16 + m;
    float bb = mlp_b[col];
#pragma unroll
    for (int r = 0; r < 8; ++r) { int row = r + 8 * hi; lt[row * 64 + col] = (_Float16)fmaxf(acc[nt][r] + bb, 0.f); }
  }
  __syncthreads();
  // GEMM2 (same weights), A from LDS
#pragma unroll
  for (int t = 0; t < 4; ++t) acc[t] = vzero8();
  const _Float16* lrow = (const _Float16*)lt + m * 64;
#pragma unroll
  for (int kt = 0; kt < 2; ++kt) {
    int q0 = kt * 32;
    v16h a = mk16(*(const v8h*)(lrow + q0 + kb), *(const v8h*)(lrow + q0 + kb + 16));
    const _Float16* wp = mlpf + (size_t)kt * 4 * 512 + (size_t)lane * 16;
#pragma unroll
    for (int nt = 0; nt < 4; ++nt)
      acc[nt] = wmma16(a, ld16(wp + (size_t)nt * 512), acc[nt]);
  }
  __syncthreads();
#pragma unroll
  for (int nt = 0; nt < 4; ++nt) {
    int col = nt * 16 + m;
    float bb = mlp_b[col];
#pragma unroll
    for (int r = 0; r < 8; ++r) { int row = r + 8 * hi; lt[row * 64 + col] = (_Float16)fmaxf(acc[nt][r] + bb, 0.f); }
  }
  __syncthreads();
  // GEMM3: heads, N=80 ([cls(2) | reg(76) | pad(2)])
  v8f acc3[5];
#pragma unroll
  for (int t = 0; t < 5; ++t) acc3[t] = vzero8();
#pragma unroll
  for (int kt = 0; kt < 2; ++kt) {
    int q0 = kt * 32;
    v16h a = mk16(*(const v8h*)(lrow + q0 + kb), *(const v8h*)(lrow + q0 + kb + 16));
    const _Float16* wp = headf + (size_t)kt * 5 * 512 + (size_t)lane * 16;
#pragma unroll
    for (int nt = 0; nt < 5; ++nt)
      acc3[nt] = wmma16(a, ld16(wp + (size_t)nt * 512), acc3[nt]);
  }
#pragma unroll
  for (int nt = 0; nt < 5; ++nt) {
    int col = nt * 16 + m;
    float bb = whead_b[col];
#pragma unroll
    for (int r = 0; r < 8; ++r) { int row = r + 8 * hi; hd[(size_t)(n0 + row) * 80 + col] = acc3[nt][r] + bb; }
  }
}

// ---------------- per-(b,p) geometry, pred assembly, prior update -----------------------------
__global__ void k_head(const float* __restrict__ hd, const float* __restrict__ pcur,
                       float* __restrict__ out, float* __restrict__ pnext,
                       float* __restrict__ pfnext, int write_next) {
  int n = blockIdx.x * 128 + threadIdx.x;
  if (n >= BP) return;
  const float* h = hd + (size_t)n * 80;
  float pr2 = pcur[(size_t)n * 78 + 2];
  float pr3 = pcur[(size_t)n * 78 + 3];
  float pr4 = pcur[(size_t)n * 78 + 4];
  float p2 = pr2 + h[2], p3 = pr3 + h[3], p4 = pr4 + h[4], p5 = h[5];
  float t = tanf(p4 * PI_F + 1e-5f);
  float a = p3 * (float)(IMG_W_ - 1);
  float binv = (float)IMG_H_ / t;
  float winv = 1.0f / (float)(IMG_W_ - 1);
  float* o = out + (size_t)n * 78;
  o[0] = h[0]; o[1] = h[1]; o[2] = p2; o[3] = p3; o[4] = p4; o[5] = p5;
  for (int j = 0; j < NPT; ++j) {
    float g = (a + ((float)j * (1.f / 71.f) - p2) * binv) * winv;  // 1 - prior_ys[j] = j/71
    o[6 + j] = g + h[6 + j];
  }
  if (write_next) {
    pnext[(size_t)n * 78 + 2] = p2;
    pnext[(size_t)n * 78 + 3] = p3;
    pnext[(size_t)n * 78 + 4] = p4;
    for (int s = 0; s < S_; ++s) {
      int j = (s * 71) / 35;
      float g = (a + ((float)j * (1.f / 71.f) - p2) * binv) * winv;
      pfnext[(size_t)n * S_ + s] = g;
    }
  }
}

// ---------------- setup kernels ----------------------------------------------------------------
__global__ void k_init_priors(const float* __restrict__ pri0, const float* __restrict__ pf0,
                              float* __restrict__ pcur, float* __restrict__ pfcur) {
  int idx = blockIdx.x * 256 + threadIdx.x;
  if (idx < BP * 78) pcur[idx] = pri0[idx % (P_ * 78)];
  if (idx < BP * S_) pfcur[idx] = pf0[idx % (P_ * S_)];
}

__global__ void k_cast(const float* __restrict__ src, _Float16* __restrict__ dst, int n) {
  int i = blockIdx.x * 256 + threadIdx.x;
  if (i < n) dst[i] = (_Float16)src[i];
}

__global__ void k_repack_conv(const float* __restrict__ src, _Float16* __restrict__ dst, int Cin) {
  int idx = blockIdx.x * 256 + threadIdx.x;
  int total = 9 * Cin * 64;
  if (idx >= total) return;
  int co = idx & 63;
  int q = idx >> 6;
  int kk = q / Cin;
  int ci = q - kk * Cin;
  dst[idx] = (_Float16)src[(size_t)(co * Cin + ci) * 9 + kk];   // src (64, Cin, 9) -> plain (9*Cin, 64)
}

__global__ void k_repack_kv(const float* __restrict__ fkey, const float* __restrict__ fval,
                            _Float16* __restrict__ dst) {
  int idx = blockIdx.x * 256 + threadIdx.x;   // plain (64, 128): dst[c*128+o]
  if (idx >= 64 * 128) return;
  int o = idx & 127;
  int c = idx >> 7;
  dst[idx] = (_Float16)(o < 64 ? fkey[o * 64 + c] : fval[(o - 64) * 64 + c]);
}

__global__ void k_repack_head(const float* __restrict__ cls_w, const float* __restrict__ cls_b,
                              const float* __restrict__ reg_w, const float* __restrict__ reg_b,
                              _Float16* __restrict__ wdst, float* __restrict__ bdst) {
  int idx = blockIdx.x * 256 + threadIdx.x;   // plain (64, 80)
  if (idx < 64 * 80) {
    int nn = idx % 80;
    int k = idx / 80;
    float v = 0.f;
    if (nn < 2) v = cls_w[k * 2 + nn];
    else if (nn < 78) v = reg_w[k * 76 + (nn - 2)];
    wdst[idx] = (_Float16)v;
  }
  if (idx < 80) {
    float v = 0.f;
    if (idx < 2) v = cls_b[idx];
    else if (idx < 78) v = reg_b[idx - 2];
    bdst[idx] = v;
  }
}

// plain f16 K x Npad (row-major) -> fragment-major: dst[((kt*ntiles+nt)*32+lane)*16+e]
// B-layout: lane holds col = nt*16 + (lane&15); K = kt*32 + (lane>>4)*8 + e + (e>=8 ? 8 : 0) (table order)
__global__ void k_swizzle(const _Float16* __restrict__ src, _Float16* __restrict__ dst,
                          int Kdim, int Npad) {
  int idx = blockIdx.x * 256 + threadIdx.x;
  int ntiles = Npad >> 4;
  int total = (Kdim >> 5) * ntiles * 512;
  if (idx >= total) return;
  int e = idx & 15;
  int lane = (idx >> 4) & 31;
  int fr = idx >> 9;
  int nt = fr % ntiles;
  int kt = fr / ntiles;
  int q = kt * 32 + ((lane >> 4) << 3) + e + ((e >= 8) ? 8 : 0);
  int col = nt * 16 + (lane & 15);
  dst[idx] = src[(size_t)q * Npad + col];
}

// ---------------- host orchestration -----------------------------------------------------------
extern "C" void kernel_launch(void* const* d_in, const int* in_sizes, int n_in,
                              void* d_out, int out_size, void* d_ws, size_t ws_size,
                              hipStream_t stream) {
  const float* x0 = (const float*)d_in[0];
  const float* x1 = (const float*)d_in[1];
  const float* x2 = (const float*)d_in[2];
  const float* priors0 = (const float*)d_in[3];
  const float* pfeat0 = (const float*)d_in[4];
  const float* conv_w = (const float*)d_in[5];
  const float* bn_conv = (const float*)d_in[6];
  const float* cat_w[3] = {(const float*)d_in[7], (const float*)d_in[8], (const float*)d_in[9]};
  const float* bn_cat = (const float*)d_in[10];
  const float* fkey_w = (const float*)d_in[11];
  const float* bn_key = (const float*)d_in[12];
  const float* fval_w = (const float*)d_in[13];
  const float* fval_b = (const float*)d_in[14];
  const float* fq_w = (const float*)d_in[15];
  const float* fq_b = (const float*)d_in[16];
  const float* attnW_w = (const float*)d_in[17];
  const float* attnW_b = (const float*)d_in[18];
  const float* fc_w = (const float*)d_in[19];
  const float* fc_b = (const float*)d_in[20];
  const float* ln_g = (const float*)d_in[21];
  const float* ln_b = (const float*)d_in[22];
  const float* mlp_w = (const float*)d_in[23];
  const float* mlp_b = (const float*)d_in[24];
  const float* reg_w = (const float*)d_in[25];
  const float* reg_b = (const float*)d_in[26];
  const float* cls_w = (const float*)d_in[27];
  const float* cls_b = (const float*)d_in[28];
  float* out = (float*)d_out;

  char* p = (char*)d_ws;
  auto alloc = [&](size_t bytes) -> char* {
    char* r = p;
    p += (bytes + 255) & ~(size_t)255;
    return r;
  };
  _Float16* pooled[3];
  for (int i = 0; i < 3; ++i) pooled[i] = (_Float16*)alloc((size_t)BP * 36 * 64 * 2);
  _Float16* cat_h = (_Float16*)alloc((size_t)BP * 36 * 192 * 2);
  _Float16* catout = (_Float16*)alloc((size_t)BP * 64 * 36 * 2);
  int cins[3] = {64, 128, 192};
  // plain f16 weights (repack staging)
  _Float16* conv_wh[3];
  for (int i = 0; i < 3; ++i) conv_wh[i] = (_Float16*)alloc((size_t)576 * 64 * 2);
  _Float16* cat_wh[3];
  for (int i = 0; i < 3; ++i) cat_wh[i] = (_Float16*)alloc((size_t)9 * cins[i] * 64 * 2);
  _Float16* fc_wh = (_Float16*)alloc((size_t)2304 * 64 * 2);
  _Float16* wkv = (_Float16*)alloc((size_t)64 * 128 * 2);
  _Float16* mlp_wh = (_Float16*)alloc((size_t)64 * 64 * 2);
  _Float16* whead = (_Float16*)alloc((size_t)64 * 80 * 2);
  // fragment-swizzled weights (consumed by GEMMs as contiguous 32B per lane)
  _Float16* conv_wf[3];
  for (int i = 0; i < 3; ++i) conv_wf[i] = (_Float16*)alloc((size_t)576 * 64 * 2);
  _Float16* cat_wf[3];
  for (int i = 0; i < 3; ++i) cat_wf[i] = (_Float16*)alloc((size_t)9 * cins[i] * 64 * 2);
  _Float16* fc_wf = (_Float16*)alloc((size_t)2304 * 64 * 2);
  _Float16* kv_wf = (_Float16*)alloc((size_t)64 * 128 * 2);
  _Float16* mlp_wf = (_Float16*)alloc((size_t)64 * 64 * 2);
  _Float16* head_wf = (_Float16*)alloc((size_t)64 * 80 * 2);
  float* whead_b = (float*)alloc(80 * 4);
  float* roi_f = (float*)alloc((size_t)BP * 64 * 4);
  _Float16* query_h = (_Float16*)alloc((size_t)BP * 64 * 2);
  _Float16* roi2_h = (_Float16*)alloc((size_t)BP * 64 * 2);
  _Float16* key_t = (_Float16*)alloc((size_t)B_ * 256 * 64 * 2);
  _Float16* value_t = (_Float16*)alloc((size_t)B_ * 64 * 256 * 2);
  _Float16* sim_h = (_Float16*)alloc((size_t)B_ * P_ * 256 * 2);
  float* hd = (float*)alloc((size_t)BP * 80 * 4);
  float* pbuf[2];
  float* pfbuf[2];
  for (int i = 0; i < 2; ++i) {
    pbuf[i] = (float*)alloc((size_t)BP * 78 * 4);
    pfbuf[i] = (float*)alloc((size_t)BP * 36 * 4);
  }

  // repack to plain f16 K x Npad
  for (int i = 0; i < 3; ++i)
    k_repack_conv<<<(9 * 64 * 64 + 255) / 256, 256, 0, stream>>>(conv_w + (size_t)i * 64 * 64 * 9, conv_wh[i], 64);
  for (int s = 0; s < 3; ++s)
    k_repack_conv<<<(9 * cins[s] * 64 + 255) / 256, 256, 0, stream>>>(cat_w[s], cat_wh[s], cins[s]);
  k_cast<<<(2304 * 64 + 255) / 256, 256, 0, stream>>>(fc_w, fc_wh, 2304 * 64);
  k_cast<<<(64 * 64 + 255) / 256, 256, 0, stream>>>(mlp_w, mlp_wh, 64 * 64);
  k_repack_kv<<<(64 * 128 + 255) / 256, 256, 0, stream>>>(fkey_w, fval_w, wkv);
  k_repack_head<<<(64 * 80 + 255) / 256, 256, 0, stream>>>(cls_w, cls_b, reg_w, reg_b, whead, whead_b);
  // fragment-swizzle all B matrices
  for (int i = 0; i < 3; ++i)
    k_swizzle<<<(9 * 64 * 64 + 255) / 256, 256, 0, stream>>>(conv_wh[i], conv_wf[i], 576, 64);
  for (int s = 0; s < 3; ++s)
    k_swizzle<<<(9 * cins[s] * 64 + 255) / 256, 256, 0, stream>>>(cat_wh[s], cat_wf[s], 9 * cins[s], 64);
  k_swizzle<<<(2304 * 64 + 255) / 256, 256, 0, stream>>>(fc_wh, fc_wf, 2304, 64);
  k_swizzle<<<(64 * 128 + 255) / 256, 256, 0, stream>>>(wkv, kv_wf, 64, 128);
  k_swizzle<<<(64 * 64 + 255) / 256, 256, 0, stream>>>(mlp_wh, mlp_wf, 64, 64);
  k_swizzle<<<(64 * 80 + 255) / 256, 256, 0, stream>>>(whead, head_wf, 64, 80);
  k_init_priors<<<(BP * 78 + 255) / 256, 256, 0, stream>>>(priors0, pfeat0, pbuf[0], pfbuf[0]);

  const float* feats[3] = {x2, x1, x0};
  int hs[3] = {10, 20, 40};
  int wsz[3] = {25, 50, 100};
  int cur = 0;
  for (int st = 0; st < 3; ++st) {
    const float* fmap = feats[st];
    int h = hs[st], w = wsz[st];
    k_pool<<<BP * 36, 64, 0, stream>>>(fmap, h, w, pfbuf[cur], pooled[st]);
    for (int i = 0; i <= st; ++i)
      k_conv<<<dim3(BP / 2, 3), 32, 0, stream>>>(pooled[i], 64 / 32, 64, conv_wf[i], bn_conv + i * 256,
                                                 cat_h, 0, i * 64, 192);
    k_conv<<<dim3(BP / 2, 3), 32, 0, stream>>>(cat_h, cins[st] / 32, 192, cat_wf[st], bn_cat + st * 256,
                                               catout, 1, 0, 0);
    k_fc<<<BP / 32, 32, 0, stream>>>(catout, fc_wf, fc_b, ln_g, ln_b, fq_w, fq_b, roi_f, query_h);
    k_kv<<<dim3(16, B_), 32, 0, stream>>>(fmap, h, w, kv_wf, bn_key, fval_b, key_t, value_t);
    k_attn1<<<dim3(12, B_), 128, 0, stream>>>(query_h, key_t, sim_h);
    k_attn2<<<dim3(12, B_), 32, 0, stream>>>(sim_h, value_t, roi_f, attnW_w, attnW_b, roi2_h);
    k_mlp<<<BP / 16, 32, 0, stream>>>(roi2_h, mlp_wf, mlp_b, head_wf, whead_b, hd);
    int wn = (st < 2) ? 1 : 0;
    k_head<<<(BP + 127) / 128, 128, 0, stream>>>(hd, pbuf[cur], out + (size_t)st * BP * 78,
                                                 pbuf[1 - cur], pfbuf[1 - cur], wn);
    cur = 1 - cur;
  }
}
